// AGNN_5463198401201
// MI455X (gfx1250) — matrix operand order repaired and checked by reading, not compiled, not run on hardware
//
#include <hip/hip_runtime.h>
#include <hip/hip_bf16.h>

// ---------------------------------------------------------------------------
// AGNN pipeline for MI455X (gfx1250, wave32, WMMA f16 16x16x32, f32 accum)
// B=8 N=128 C=128 G=2 H=4 DK=32
// Weights are pre-packed into WMMA B-fragment tiles (32 lanes x 16 halves)
// so every B-fragment load is one contiguous 32B read per lane.
// ---------------------------------------------------------------------------

#define B_  8
#define N_  128
#define C_  128
#define DK_ 32
#define BN_SCALE 0.99999500003749969f   // 1/sqrt(1+1e-5)

typedef __attribute__((ext_vector_type(16))) _Float16 v16h;
typedef __attribute__((ext_vector_type(8)))  _Float16 v8h;
typedef __attribute__((ext_vector_type(8)))  float    v8f;

static __device__ __forceinline__ int lane32() { return (int)(threadIdx.x & 31u); }

static __device__ __forceinline__ v8f vzero() {
  v8f z = {0.f, 0.f, 0.f, 0.f, 0.f, 0.f, 0.f, 0.f};
  return z;
}

// A fragment (16x32 f16, row-major source, leading dim ldk halves).
// Layout: lane m=l&15 ; lanes16-31 shift K by +16 ; halves 0-15 -> K 0..15
// (relative, contiguous). Loaded as two 16B vectors.
static __device__ __forceinline__ v16h frag_a(const _Float16* __restrict__ base, int ldk) {
  const int l = lane32();
  const _Float16* row = base + (l & 15) * ldk + ((l & 16) ? 16 : 0);
  v8h lo = *(const v8h*)(row);
  v8h hi = *(const v8h*)(row + 8);
  v16h a;
#pragma unroll
  for (int h = 0; h < 8; ++h) { a[h] = lo[h]; a[8 + h] = hi[h]; }
  return a;
}

// Packed B fragment: tile holds 32 lanes x 16 halves, one 32B read per lane.
static __device__ __forceinline__ v16h frag_bp(const _Float16* __restrict__ tile) {
  return ((const v16h*)tile)[lane32()];
}

// B fragment where logical B[k][n] = src[n*ld + k] (transposed access, e.g. K^T).
// 16 contiguous halves per lane -> single 32B read.
static __device__ __forceinline__ v16h frag_bT(const _Float16* __restrict__ base, int ld) {
  const int l = lane32();
  return *(const v16h*)(base + (l & 15) * ld + ((l & 16) ? 16 : 0));
}

static __device__ __forceinline__ v8f wmma_f16(v16h a, v16h b, v8f c) {
  return __builtin_amdgcn_wmma_f32_16x16x32_f16(false, a, false, b, (short)0, c,
                                                false, false);
}

// D layout: lane -> col n=l&15, row block mb=(l&16)?8:0, vgpr r -> row mb+r.
static __device__ __forceinline__ void store_d_f16(_Float16* base, int ldn, v8f d) {
  const int l = lane32();
  const int n = l & 15, mb = (l & 16) ? 8 : 0;
#pragma unroll
  for (int r = 0; r < 8; ++r) base[(mb + r) * ldn + n] = (_Float16)d[r];
}

static __device__ __forceinline__ void store_d_f16_bnlrelu(_Float16* base, int ldn, v8f d) {
  const int l = lane32();
  const int n = l & 15, mb = (l & 16) ? 8 : 0;
#pragma unroll
  for (int r = 0; r < 8; ++r) {
    float x = d[r] * BN_SCALE;
    x = x > 0.f ? x : 0.01f * x;
    base[(mb + r) * ldn + n] = (_Float16)x;
  }
}

static __device__ __forceinline__ void store_d_f32_scaled(float* base, int ldn, v8f d,
                                                          float s) {
  const int l = lane32();
  const int n = l & 15, mb = (l & 16) ? 8 : 0;
#pragma unroll
  for (int r = 0; r < 8; ++r) base[(mb + r) * ldn + n] = d[r] * s;
}

// ---------------------------------------------------------------------------
// Weight packing: f32 [K][Nd] row-major -> f16 B-fragment tiles.
// grid = (K/32)*NT blocks of 32 threads; block = one (kt,nt) tile.
// ---------------------------------------------------------------------------
__global__ __launch_bounds__(32) void k_pack_w(const float* __restrict__ src,
                                               _Float16* __restrict__ dst,
                                               int NT, int Nd) {
  const int l = (int)threadIdx.x;
  const int kt = (int)blockIdx.x / NT, nt = (int)blockIdx.x % NT;
  const int k0 = kt * 32 + ((l & 16) ? 16 : 0);
  const int n = nt * 16 + (l & 15);
  v16h v;
#pragma unroll
  for (int h = 0; h < 16; ++h) v[h] = (_Float16)src[(size_t)(k0 + h) * Nd + n];
  ((v16h*)(dst + (size_t)blockIdx.x * 512))[l] = v;
}

// Activation packing: f16 [K][Nd] row-major -> B-fragment tiles, per matrix.
// grid = nmat * (K/32)*NT
__global__ __launch_bounds__(32) void k_pack_act(const _Float16* __restrict__ src,
                                                 _Float16* __restrict__ dst,
                                                 int NT, int Nd, int tilesPerMat,
                                                 int matStride) {
  const int l = (int)threadIdx.x;
  const int mat = (int)blockIdx.x / tilesPerMat;
  const int tid = (int)blockIdx.x % tilesPerMat;
  const int kt = tid / NT, nt = tid % NT;
  const int k0 = kt * 32 + ((l & 16) ? 16 : 0);
  const int n = nt * 16 + (l & 15);
  const _Float16* s = src + (size_t)mat * matStride;
  v16h v;
#pragma unroll
  for (int h = 0; h < 16; ++h) v[h] = s[(size_t)(k0 + h) * Nd + n];
  ((v16h*)(dst + (size_t)mat * matStride + (size_t)tid * 512))[l] = v;
}

// ---------------------------------------------------------------------------
// fp32 -> f16 conversion
// ---------------------------------------------------------------------------
__global__ __launch_bounds__(256) void k_cvt(const float* __restrict__ s,
                                             _Float16* __restrict__ d, int n) {
  int i = blockIdx.x * 256 + threadIdx.x;
  if (i < n) d[i] = (_Float16)s[i];
}

// ---------------------------------------------------------------------------
// Fused PointSimilarity MLP: e_raw[b,i,j] = sigmoid(MLP((vp_i - vp_j)^2))
// grid = B*N*2 blocks, 128 threads (4 waves); each wave: 16 pairs.
// w1p: packed tiles, K=128 (kt 0..3), NT=16. w2p: K=256 (kt 0..7), NT=8.
// ---------------------------------------------------------------------------
__global__ __launch_bounds__(128) void k_psim_eraw(
    const float* __restrict__ vp, const _Float16* __restrict__ w1p,
    const _Float16* __restrict__ w2p, const float* __restrict__ w3,
    const float* __restrict__ b3, float* __restrict__ eraw) {
  __shared__ _Float16 sm[4][4096];  // 8KB per wave, reused across layers
  const int wave = threadIdx.x >> 5;
  const int l = lane32();
  const int blk = blockIdx.x;
  const int jblk = blk & 1;
  const int i = (blk >> 1) & (N_ - 1);
  const int b = blk >> 8;
  const int jbase = jblk * 64 + wave * 16;
  _Float16* S = &sm[wave][0];

  // ---- stage sim features f16 [16 pairs][128 chan] ----
  const float4 vi4 = *(const float4*)(vp + (b * N_ + i) * C_ + l * 4);
#pragma unroll
  for (int p = 0; p < 16; ++p) {
    const float4 vj4 = *(const float4*)(vp + (b * N_ + jbase + p) * C_ + l * 4);
    float d0 = vi4.x - vj4.x, d1 = vi4.y - vj4.y;
    float d2 = vi4.z - vj4.z, d3 = vi4.w - vj4.w;
    v8h* dst = (v8h*)(S + p * 128 + l * 4);  // 4 halves; write via scalar
    _Float16* ds = (_Float16*)dst;
    ds[0] = (_Float16)(d0 * d0); ds[1] = (_Float16)(d1 * d1);
    ds[2] = (_Float16)(d2 * d2); ds[3] = (_Float16)(d3 * d3);
  }

  // ---- layer1: [16x128] x [128x256] ----
  v8f acc1[16];
#pragma unroll
  for (int t = 0; t < 16; ++t) acc1[t] = vzero();
  for (int k0 = 0; k0 < 4; ++k0) {
    v16h a = frag_a(S + k0 * 32, 128);
#pragma unroll
    for (int t = 0; t < 16; ++t)
      acc1[t] = wmma_f16(a, frag_bp(w1p + (k0 * 16 + t) * 512), acc1[t]);
  }
#pragma unroll
  for (int t = 0; t < 16; ++t) store_d_f16_bnlrelu(S + t * 16, 256, acc1[t]);

  // ---- layer2: [16x256] x [256x128] ----
  v8f acc2[8];
#pragma unroll
  for (int t = 0; t < 8; ++t) acc2[t] = vzero();
  for (int k0 = 0; k0 < 8; ++k0) {
    v16h a = frag_a(S + k0 * 32, 256);
#pragma unroll
    for (int t = 0; t < 8; ++t)
      acc2[t] = wmma_f16(a, frag_bp(w2p + (k0 * 8 + t) * 512), acc2[t]);
  }
#pragma unroll
  for (int t = 0; t < 8; ++t) store_d_f16_bnlrelu(S + t * 16, 128, acc2[t]);

  // ---- layer3: sigmoid(h2 . w3 + b3) ----
  float w34[4];
#pragma unroll
  for (int q = 0; q < 4; ++q) w34[q] = w3[l * 4 + q];
  const float bb3 = b3[0];
  for (int p = 0; p < 16; ++p) {
    float s = 0.f;
#pragma unroll
    for (int q = 0; q < 4; ++q) s += (float)S[p * 128 + l * 4 + q] * w34[q];
#pragma unroll
    for (int off = 16; off > 0; off >>= 1) s += __shfl_xor(s, off, 32);
    if (l == 0) {
      float x = s + bb3;
      eraw[(b * N_ + i) * N_ + jbase + p] = 1.f / (1.f + __expf(-x));
    }
  }
}

// ---------------------------------------------------------------------------
// psim post: diag mask, ep_sum, top-k, l1norm, +eye+1e-6, row-normalize
// grid = B*N, 128 threads (one per j).
// ---------------------------------------------------------------------------
__global__ __launch_bounds__(128) void k_psim_post(const float* __restrict__ eraw,
                                                   const float* __restrict__ ep_last,
                                                   float* __restrict__ ep_out, int kval) {
  __shared__ float ev[128];
  __shared__ float red[128];
  const int t = threadIdx.x;
  const int i = blockIdx.x & (N_ - 1);
  const int b = blockIdx.x >> 7;
  const int row = (b * N_ + i) * N_;

  float epl = ep_last[row + t];
  if (t == i) epl = 0.f;
  red[t] = epl; __syncthreads();
  for (int s = 64; s > 0; s >>= 1) { if (t < s) red[t] += red[t + s]; __syncthreads(); }
  const float ep_sum = red[0]; __syncthreads();

  float e = eraw[row + t] * epl;
  if (kval > 0) {  // keep top-k per row (rank by count-greater)
    ev[t] = e; __syncthreads();
    int cnt = 0;
    for (int j = 0; j < 128; ++j) cnt += (ev[j] > e) ? 1 : 0;
    if (cnt >= kval) e = 0.f;
    __syncthreads();
  }
  red[t] = fabsf(e); __syncthreads();
  for (int s = 64; s > 0; s >>= 1) { if (t < s) red[t] += red[t + s]; __syncthreads(); }
  const float l1 = red[0]; __syncthreads();

  e = e / fmaxf(l1, 1e-12f) * ep_sum;
  e += ((t == i) ? 1.f : 0.f) + 1e-6f;
  red[t] = e; __syncthreads();
  for (int s = 64; s > 0; s >>= 1) { if (t < s) red[t] += red[t + s]; __syncthreads(); }
  ep_out[row + t] = e / red[0];
}

// ---------------------------------------------------------------------------
// MHA projections: Q = vp@wq, K = vp@wk (f16 out). grid B*8, 128 threads.
// wq/wk packed: K=128 (kt 0..3), NT=8.
// ---------------------------------------------------------------------------
__global__ __launch_bounds__(128) void k_mha_qk(const _Float16* __restrict__ vph,
                                                const _Float16* __restrict__ wqp,
                                                const _Float16* __restrict__ wkp,
                                                _Float16* __restrict__ Qh,
                                                _Float16* __restrict__ Kh) {
  const int wave = threadIdx.x >> 5;
  const int it = blockIdx.x & 7;
  const int b = blockIdx.x >> 3;
  const int i0 = it * 16;
  const _Float16* A0 = vph + (b * N_ + i0) * C_;
  v8f aq[2] = {vzero(), vzero()}, ak[2] = {vzero(), vzero()};
  for (int k0 = 0; k0 < 4; ++k0) {
    v16h a = frag_a(A0 + k0 * 32, 128);
#pragma unroll
    for (int t = 0; t < 2; ++t) {
      const int nt = wave * 2 + t;
      aq[t] = wmma_f16(a, frag_bp(wqp + (k0 * 8 + nt) * 512), aq[t]);
      ak[t] = wmma_f16(a, frag_bp(wkp + (k0 * 8 + nt) * 512), ak[t]);
    }
  }
#pragma unroll
  for (int t = 0; t < 2; ++t) {
    const int nt = wave * 2 + t;
    store_d_f16(Qh + (b * N_ + i0) * C_ + nt * 16, 128, aq[t]);
    store_d_f16(Kh + (b * N_ + i0) * C_ + nt * 16, 128, ak[t]);
  }
}

// ---------------------------------------------------------------------------
// MHA scores + softmax + head mean. grid B*8, 128 threads (wave == head).
// ---------------------------------------------------------------------------
__global__ __launch_bounds__(128) void k_mha_attn(const _Float16* __restrict__ Qh,
                                                  const _Float16* __restrict__ Kh,
                                                  float* __restrict__ attn) {
  __shared__ float S[4][16][128];
  const int h = threadIdx.x >> 5;  // head
  const int it = blockIdx.x & 7;
  const int b = blockIdx.x >> 3;
  const int i0 = it * 16;
  const float sc = 0.17677669529663687f;  // 1/sqrt(DK)

  v16h a = frag_a(Qh + (b * N_ + i0) * C_ + h * DK_, 128);  // K-dim = 32, one step
  for (int jt = 0; jt < 8; ++jt) {
    v16h bb = frag_bT(Kh + (b * N_ + jt * 16) * C_ + h * DK_, 128);
    v8f acc = wmma_f16(a, bb, vzero());
    store_d_f32_scaled(&S[h][0][jt * 16], 128, acc, sc);
  }
  __syncthreads();
  const int t = threadIdx.x;
  if (t < 64) {  // softmax over key dim; one (head,row) per thread
    float* r = &S[t >> 4][t & 15][0];
    float m = -1e30f;
    for (int j = 0; j < 128; ++j) m = fmaxf(m, r[j]);
    float sum = 0.f;
    for (int j = 0; j < 128; ++j) { float ee = __expf(r[j] - m); r[j] = ee; sum += ee; }
    float inv = 1.f / sum;
    for (int j = 0; j < 128; ++j) r[j] *= inv;
  }
  __syncthreads();
  for (int idx = t; idx < 16 * 128; idx += 128) {
    const int m = idx >> 7, j = idx & 127;
    attn[(b * N_ + i0 + m) * N_ + j] =
        0.25f * (S[0][m][j] + S[1][m][j] + S[2][m][j] + S[3][m][j]);
  }
}

// ---------------------------------------------------------------------------
// edge = l1norm(ep*attn, diag zeroed) (f16);  nf[:, :128] = vp (f16).
// grid B*N, 128 threads.
// ---------------------------------------------------------------------------
__global__ __launch_bounds__(128) void k_edge_prep(const float* __restrict__ ep,
                                                   const float* __restrict__ attn,
                                                   const _Float16* __restrict__ vph,
                                                   _Float16* __restrict__ edgeh,
                                                   _Float16* __restrict__ nfh) {
  __shared__ float red[128];
  const int t = threadIdx.x;
  const int i = blockIdx.x & (N_ - 1);
  const int b = blockIdx.x >> 7;
  const int row = (b * N_ + i) * N_;
  float e = ep[row + t] * attn[row + t];
  if (t == i) e = 0.f;
  red[t] = fabsf(e); __syncthreads();
  for (int s = 64; s > 0; s >>= 1) { if (t < s) red[t] += red[t + s]; __syncthreads(); }
  const float inv = 1.f / fmaxf(red[0], 1e-12f);
  edgeh[row + t] = (_Float16)(e * inv);
  nfh[(b * N_ + i) * 256 + t] = vph[(b * N_ + i) * C_ + t];
}

// ---------------------------------------------------------------------------
// aggr = edge @ vp  -> nf[:, 128:256] (f16). grid B*8, 128 threads.
// vpackp: per-batch packed B tiles of vp (K=128: kt 0..3, NT=8).
// ---------------------------------------------------------------------------
__global__ __launch_bounds__(128) void k_aggr(const _Float16* __restrict__ edgeh,
                                              const _Float16* __restrict__ vpackp,
                                              _Float16* __restrict__ nfh) {
  const int wave = threadIdx.x >> 5;
  const int it = blockIdx.x & 7;
  const int b = blockIdx.x >> 3;
  const int i0 = it * 16;
  const _Float16* A0 = edgeh + (b * N_ + i0) * N_;
  const _Float16* Bp = vpackp + (size_t)b * 16384;
  v8f acc[2] = {vzero(), vzero()};
  for (int k0 = 0; k0 < 4; ++k0) {
    v16h a = frag_a(A0 + k0 * 32, 128);
#pragma unroll
    for (int t = 0; t < 2; ++t) {
      const int nt = wave * 2 + t;
      acc[t] = wmma_f16(a, frag_bp(Bp + (k0 * 8 + nt) * 512), acc[t]);
    }
  }
#pragma unroll
  for (int t = 0; t < 2; ++t) {
    const int nt = wave * 2 + t;
    store_d_f16(nfh + (b * N_ + i0) * 256 + 128 + nt * 16, 256, acc[t]);
  }
}

// ---------------------------------------------------------------------------
// D2P MLP: vp_next = lrelu(bn(lrelu(bn(nf@w1))@w2)). grid B*8, 128 threads.
// w1p: K=256 (kt 0..7), NT=16. w2p: K=256 (kt 0..7), NT=8.
// ---------------------------------------------------------------------------
__global__ __launch_bounds__(128) void k_d2p_mlp(const _Float16* __restrict__ nfh,
                                                 const _Float16* __restrict__ w1p,
                                                 const _Float16* __restrict__ w2p,
                                                 float* __restrict__ vpf,
                                                 _Float16* __restrict__ vph) {
  __shared__ _Float16 h1[16 * 256];
  const int wave = threadIdx.x >> 5;
  const int it = blockIdx.x & 7;
  const int b = blockIdx.x >> 3;
  const int i0 = it * 16;
  const _Float16* A0 = nfh + (b * N_ + i0) * 256;

  v8f acc[4] = {vzero(), vzero(), vzero(), vzero()};
  for (int k0 = 0; k0 < 8; ++k0) {
    v16h a = frag_a(A0 + k0 * 32, 256);
#pragma unroll
    for (int t = 0; t < 4; ++t) {
      const int nt = wave * 4 + t;
      acc[t] = wmma_f16(a, frag_bp(w1p + (k0 * 16 + nt) * 512), acc[t]);
    }
  }
#pragma unroll
  for (int t = 0; t < 4; ++t) store_d_f16_bnlrelu(h1 + (wave * 4 + t) * 16, 256, acc[t]);
  __syncthreads();

  v8f acc2[2] = {vzero(), vzero()};
  for (int k0 = 0; k0 < 8; ++k0) {
    v16h a = frag_a(h1 + k0 * 32, 256);
#pragma unroll
    for (int t = 0; t < 2; ++t) {
      const int nt = wave * 2 + t;
      acc2[t] = wmma_f16(a, frag_bp(w2p + (k0 * 8 + nt) * 512), acc2[t]);
    }
  }
  const int l = lane32();
  const int n = l & 15, mb = (l & 16) ? 8 : 0;
#pragma unroll
  for (int t = 0; t < 2; ++t) {
    const int nt = wave * 2 + t;
#pragma unroll
    for (int r = 0; r < 8; ++r) {
      float x = acc2[t][r] * BN_SCALE;
      x = x > 0.f ? x : 0.01f * x;
      const int m = mb + r, c = nt * 16 + n;
      vpf[(b * N_ + i0 + m) * C_ + c] = x;
      vph[(b * N_ + i0 + m) * C_ + c] = (_Float16)x;
    }
  }
}

// ---------------------------------------------------------------------------
// Host orchestration
// ---------------------------------------------------------------------------
extern "C" void kernel_launch(void* const* d_in, const int* in_sizes, int n_in,
                              void* d_out, int out_size, void* d_ws, size_t ws_size,
                              hipStream_t stream) {
  (void)in_sizes; (void)n_in; (void)out_size; (void)ws_size;
  const float* vp_in  = (const float*)d_in[0];
  const float* ep0    = (const float*)d_in[1];
  const float* pre_w1 = (const float*)d_in[2];
  const float* pre_w2 = (const float*)d_in[3];
  const float* pre_w3 = (const float*)d_in[4];
  const float* pre_b3 = (const float*)d_in[5];
  const float* ps_w1  = (const float*)d_in[6];
  const float* ps_w2  = (const float*)d_in[7];
  const float* ps_w3  = (const float*)d_in[8];
  const float* ps_b3  = (const float*)d_in[9];
  const float* d2_w1  = (const float*)d_in[10];
  const float* d2_w2  = (const float*)d_in[11];
  const float* mh_wq  = (const float*)d_in[12];
  const float* mh_wk  = (const float*)d_in[13];

  const int BNN = B_ * N_ * N_;   // 131072
  const int BNC = B_ * N_ * C_;   // 131072

  char* ws = (char*)d_ws;
  size_t off = 0;
  auto alloc = [&](size_t bytes) -> char* {
    char* p = ws + off;
    off = (off + bytes + 255) & ~(size_t)255;
    return p;
  };
  float*    epA    = (float*)alloc((size_t)BNN * 4);
  float*    epB    = (float*)alloc((size_t)BNN * 4);
  float*    eraw   = (float*)alloc((size_t)BNN * 4);
  float*    attn   = (float*)alloc((size_t)BNN * 4);
  float*    vpf    = (float*)alloc((size_t)BNC * 4);
  _Float16* vph    = (_Float16*)alloc((size_t)BNC * 2);
  _Float16* vpack  = (_Float16*)alloc((size_t)BNC * 2);
  _Float16* edgeh  = (_Float16*)alloc((size_t)BNN * 2);
  _Float16* Qh     = (_Float16*)alloc((size_t)BNC * 2);
  _Float16* Kh     = (_Float16*)alloc((size_t)BNC * 2);
  _Float16* nfh    = (_Float16*)alloc((size_t)B_ * N_ * 2 * C_ * 2);
  _Float16* pre_w1p = (_Float16*)alloc(32768 * 2);   // 4*16 tiles
  _Float16* pre_w2p = (_Float16*)alloc(32768 * 2);   // 8*8 tiles
  _Float16* ps_w1p  = (_Float16*)alloc(65536 * 2);   // G * 4*16 tiles
  _Float16* ps_w2p  = (_Float16*)alloc(65536 * 2);   // G * 8*8 tiles
  _Float16* d2_w1p  = (_Float16*)alloc(131072 * 2);  // G * 8*16 tiles
  _Float16* d2_w2p  = (_Float16*)alloc(65536 * 2);   // G * 8*8 tiles
  _Float16* wqp     = (_Float16*)alloc(32768 * 2);   // G * 4*8 tiles
  _Float16* wkp     = (_Float16*)alloc(32768 * 2);

  // -------- pack all weights into B-fragment tiles (f32 -> f16) --------
  auto pack = [&](const float* s, _Float16* d, int K, int Nd) {
    k_pack_w<<<(K / 32) * (Nd / 16), 32, 0, stream>>>(s, d, Nd / 16, Nd);
  };
  pack(pre_w1, pre_w1p, 128, 256);
  pack(pre_w2, pre_w2p, 256, 128);
  for (int g = 0; g < 2; ++g) {
    pack(ps_w1 + g * 32768, ps_w1p + g * 32768, 128, 256);
    pack(ps_w2 + g * 32768, ps_w2p + g * 32768, 256, 128);
    pack(d2_w1 + g * 65536, d2_w1p + g * 65536, 256, 256);
    pack(d2_w2 + g * 32768, d2_w2p + g * 32768, 256, 128);
    pack(mh_wq + g * 16384, wqp + g * 16384, 128, 128);
    pack(mh_wk + g * 16384, wkp + g * 16384, 128, 128);
  }
  k_cvt<<<(BNC + 255) / 256, 256, 0, stream>>>(vp_in, vph, BNC);

  // ep = psim_pre(vp, ep0), no top-k
  k_psim_eraw<<<B_ * N_ * 2, 128, 0, stream>>>(vp_in, pre_w1p, pre_w2p, pre_w3, pre_b3,
                                               eraw);
  k_psim_post<<<B_ * N_, 128, 0, stream>>>(eraw, ep0, epA, 0);

  const float* ep_cur = epA;
  for (int g = 0; g < 2; ++g) {
    k_mha_qk<<<B_ * 8, 128, 0, stream>>>(vph, wqp + g * 16384, wkp + g * 16384, Qh, Kh);
    k_mha_attn<<<B_ * 8, 128, 0, stream>>>(Qh, Kh, attn);
    k_edge_prep<<<B_ * N_, 128, 0, stream>>>(ep_cur, attn, vph, edgeh, nfh);
    // pack current vp (f16 row-major) into per-batch B tiles for aggr
    k_pack_act<<<B_ * 32, 32, 0, stream>>>(vph, vpack, 8, 128, 32, 16384);
    k_aggr<<<B_ * 8, 128, 0, stream>>>(edgeh, vpack, nfh);
    k_d2p_mlp<<<B_ * 8, 128, 0, stream>>>(nfh, d2_w1p + g * 65536, d2_w2p + g * 32768,
                                          vpf, vph);
    k_psim_eraw<<<B_ * N_ * 2, 128, 0, stream>>>(vpf, ps_w1p + g * 32768,
                                                 ps_w2p + g * 32768, ps_w3 + g * 128,
                                                 ps_b3 + g, eraw);
    float* ep_next = (g == 0) ? epB : (float*)d_out;
    const int kval = (g == 0) ? 115 : 102;  // int(N*(1-0.1*(g+1)))
    k_psim_post<<<B_ * N_, 128, 0, stream>>>(eraw, ep_cur, ep_next, kval);
    ep_cur = ep_next;
  }
}
